// EGNN_31241592111734
// MI455X (gfx1250) — matrix operand order repaired and checked by reading, not compiled, not run on hardware
//
#include <hip/hip_runtime.h>

typedef __attribute__((ext_vector_type(16))) __bf16 v16bf;
typedef __attribute__((ext_vector_type(8)))  float  v8f;

#define HID 128
#define NRBF 32
#define KMSG (2*HID + NRBF)   // 288
#define KUPD (2*HID)          // 256
#define NLAYERS 4

__device__ __forceinline__ unsigned short f2bf(float f) {
    union { float f; unsigned int u; } x; x.f = f;
    unsigned int u = x.u + 0x7FFFu + ((x.u >> 16) & 1u);   // RNE
    return (unsigned short)(u >> 16);
}
__device__ __forceinline__ float bf2f(unsigned short b) {
    union { unsigned int u; float f; } x; x.u = ((unsigned int)b) << 16;
    return x.f;
}

union AF {            // one wave-lane's 16-bf16 WMMA fragment
    v16bf v;
    uint4 q[2];
    unsigned short u[16];
};

__device__ __forceinline__ v8f wmma_bf16(const AF& a, const AF& b, v8f c) {
    return __builtin_amdgcn_wmma_f32_16x16x32_bf16(false, a.v, false, b.v,
                                                   (short)0, c, false, false);
}

// A-fragment from a row-major bf16 row: elems 0-7 = K ko..ko+7, elems 8-15 = K ko+16..ko+23
__device__ __forceinline__ void load_a_row(AF& f, const unsigned short* base) {
    const uint4* p = (const uint4*)base;
    f.q[0] = p[0];
    f.q[1] = p[2];
}
// B-fragment from [N][K] bf16 layout: elems 0-7 = K ko..ko+7, elems 8-15 = K ko+16..ko+23
__device__ __forceinline__ void load_b(AF& f, const unsigned short* base) {
    const uint4* p = (const uint4*)base;
    f.q[0] = p[0];
    f.q[1] = p[2];
}

// ---------------- setup kernels ----------------

__global__ void k_zero(float* p, long n) {
    long i = (long)blockIdx.x * 256 + threadIdx.x;
    long stride = (long)gridDim.x * 256;
    for (; i < n; i += stride) p[i] = 0.f;
}

__global__ void k_embed(const int* __restrict__ z, const float* __restrict__ emb,
                        unsigned short* __restrict__ hbf, int N) {
    long idx = (long)blockIdx.x * 256 + threadIdx.x;
    if (idx >= (long)N * HID) return;
    int n = (int)(idx >> 7), c = (int)(idx & 127);
    hbf[idx] = f2bf(emb[(long)z[n] * HID + c]);
}

__global__ void k_edge(const float* __restrict__ pos, const int* __restrict__ ei,
                       const int* __restrict__ ej, float* __restrict__ dist,
                       float* __restrict__ deg, int E) {
    int e = blockIdx.x * 256 + threadIdx.x;
    if (e >= E) return;
    int i = ei[e], j = ej[e];
    float dx = pos[3*i]   - pos[3*j];
    float dy = pos[3*i+1] - pos[3*j+1];
    float dz = pos[3*i+2] - pos[3*j+2];
    dist[e] = sqrtf(dx*dx + dy*dy + dz*dz);
    atomicAdd(deg + i, 1.0f);
}

__global__ void k_invdeg(float* __restrict__ deg, int N) {
    int n = blockIdx.x * 256 + threadIdx.x;
    if (n >= N) return;
    deg[n] = 1.0f / fmaxf(deg[n], 1.0f);
}

// transpose W[l][K][128] fp32  ->  Wt[l][128][K] bf16
__global__ void k_wt(const float* __restrict__ w, unsigned short* __restrict__ wt,
                     int K, int L) {
    long idx = (long)blockIdx.x * 256 + threadIdx.x;
    long total = (long)L * K * HID;
    if (idx >= total) return;
    long l = idx / ((long)K * HID);
    long r = idx - l * (long)K * HID;
    int k = (int)(r >> 7), n = (int)(r & 127);
    wt[l * (long)K * HID + (long)n * K + k] = f2bf(w[idx]);
}

// ---------------- message kernel: m_ij = MLP([h_i, h_j, rbf]); aggr[i] += m_ij ----------------

__global__ __launch_bounds__(256) void k_message(
    const unsigned short* __restrict__ hbf, const float* __restrict__ dist,
    const int* __restrict__ ei, const int* __restrict__ ej,
    const unsigned short* __restrict__ w1t, const unsigned short* __restrict__ w2t,
    const float* __restrict__ b1g, const float* __restrict__ b2g,
    float* __restrict__ aggr, int E)
{
    __shared__ unsigned short s_w1[KMSG * HID];    // 73728 B
    __shared__ unsigned short s_w2[HID * HID];     // 32768 B
    __shared__ unsigned short s_scr[8][16 * HID];  // 32768 B
    __shared__ int            s_tgt[8][16];

    // stage weights
    {
        const uint4* src = (const uint4*)w1t;  uint4* dst = (uint4*)s_w1;
        for (int t = threadIdx.x; t < KMSG*HID/8; t += 256) dst[t] = src[t];
        src = (const uint4*)w2t;  dst = (uint4*)s_w2;
        for (int t = threadIdx.x; t < HID*HID/8; t += 256) dst[t] = src[t];
    }
    __syncthreads();

    int wave = threadIdx.x >> 5, lane = threadIdx.x & 31;
    int half = lane >> 4, m = lane & 15;
    int ko = half * 8, h16 = half * 16;
    int tile = blockIdx.x * 8 + wave;
    if (tile * 16 >= E) return;

    int e = tile * 16 + m;
    int ec = (e < E) ? e : (E - 1);
    int ni = ei[ec], nj = ej[ec];
    float d = dist[ec];
    if (!half && e < E) s_tgt[wave][m] = ni;
    unsigned short* scr = s_scr[wave];

    v8f acc[8];
#pragma unroll
    for (int nt = 0; nt < 8; ++nt) acc[nt] = v8f{0.f,0.f,0.f,0.f,0.f,0.f,0.f,0.f};

    // GEMM1: (16 x 288) @ (288 x 128)  -- batch-preload B frags per K-chunk
#pragma unroll
    for (int c = 0; c < 8; ++c) {
        AF a;
        const unsigned short* rowp = (c < 4)
            ? hbf + (long)ni * HID + c * 32
            : hbf + (long)nj * HID + (c - 4) * 32;
        load_a_row(a, rowp + ko);
        AF b[8];
#pragma unroll
        for (int nt = 0; nt < 8; ++nt)
            load_b(b[nt], s_w1 + (nt*16 + m) * KMSG + c*32 + ko);
#pragma unroll
        for (int nt = 0; nt < 8; ++nt)
            acc[nt] = wmma_bf16(a, b[nt], acc[nt]);
    }
    {   // RBF chunk (K = 256..287)
        AF a;
#pragma unroll
        for (int t = 0; t < 16; ++t) {
            int k = (t < 8) ? (ko + t) : (ko + 8 + t);
            float cen = (float)k * (10.0f / 31.0f);
            float dd = d - cen;
            a.u[t] = f2bf(__expf(-3.2f * dd * dd));
        }
        AF b[8];
#pragma unroll
        for (int nt = 0; nt < 8; ++nt)
            load_b(b[nt], s_w1 + (nt*16 + m) * KMSG + 256 + ko);
#pragma unroll
        for (int nt = 0; nt < 8; ++nt)
            acc[nt] = wmma_bf16(a, b[nt], acc[nt]);
    }

    // bias + relu -> LDS scratch (transpose D-layout -> A-layout)
#pragma unroll
    for (int nt = 0; nt < 8; ++nt) {
        int ncol = nt*16 + m;
        float bias = b1g[ncol];
#pragma unroll
        for (int r = 0; r < 8; ++r) {
            float v = fmaxf(acc[nt][r] + bias, 0.f);
            int row = half ? (r + 8) : r;
            scr[row * HID + ncol] = f2bf(v);
        }
    }

    // GEMM2: (16 x 128) @ (128 x 128)
    v8f acc2[8];
#pragma unroll
    for (int nt = 0; nt < 8; ++nt) acc2[nt] = v8f{0.f,0.f,0.f,0.f,0.f,0.f,0.f,0.f};
#pragma unroll
    for (int c2 = 0; c2 < 4; ++c2) {
        AF a; load_a_row(a, scr + m * HID + c2*32 + ko);
        AF b[8];
#pragma unroll
        for (int nt = 0; nt < 8; ++nt)
            load_b(b[nt], s_w2 + (nt*16 + m) * HID + c2*32 + ko);
#pragma unroll
        for (int nt = 0; nt < 8; ++nt)
            acc2[nt] = wmma_bf16(a, b[nt], acc2[nt]);
    }

    // bias + atomic scatter-add into aggr[target]
#pragma unroll
    for (int nt = 0; nt < 8; ++nt) {
        int ncol = nt*16 + m;
        float bias = b2g[ncol];
#pragma unroll
        for (int r = 0; r < 8; ++r) {
            int row = half ? (r + 8) : r;
            int e2 = tile * 16 + row;
            if (e2 < E) {
                int node = s_tgt[wave][row];
                atomicAdd(&aggr[(long)node * HID + ncol], acc2[nt][r] + bias);
            }
        }
    }
}

// ---------------- update kernel: h = MLP([h, aggr*inv_deg]) ----------------

__global__ __launch_bounds__(256) void k_update(
    unsigned short* __restrict__ hbf, const float* __restrict__ aggr,
    const float* __restrict__ invdeg,
    const unsigned short* __restrict__ w1t, const unsigned short* __restrict__ w2t,
    const float* __restrict__ b1g, const float* __restrict__ b2g, int N)
{
    __shared__ unsigned short s_w1[KUPD * HID];    // 65536 B
    __shared__ unsigned short s_w2[HID * HID];     // 32768 B
    __shared__ unsigned short s_scr[8][16 * HID];  // 32768 B

    {
        const uint4* src = (const uint4*)w1t;  uint4* dst = (uint4*)s_w1;
        for (int t = threadIdx.x; t < KUPD*HID/8; t += 256) dst[t] = src[t];
        src = (const uint4*)w2t;  dst = (uint4*)s_w2;
        for (int t = threadIdx.x; t < HID*HID/8; t += 256) dst[t] = src[t];
    }
    __syncthreads();

    int wave = threadIdx.x >> 5, lane = threadIdx.x & 31;
    int half = lane >> 4, m = lane & 15;
    int ko = half * 8, h16 = half * 16;
    int tile = blockIdx.x * 8 + wave;
    if (tile * 16 >= N) return;

    int node = tile * 16 + m;
    int nc = (node < N) ? node : (N - 1);
    float invd = invdeg[nc];
    unsigned short* scr = s_scr[wave];

    v8f acc[8];
#pragma unroll
    for (int nt = 0; nt < 8; ++nt) acc[nt] = v8f{0.f,0.f,0.f,0.f,0.f,0.f,0.f,0.f};

    // GEMM1: (16 x 256) @ (256 x 128)
#pragma unroll
    for (int c = 0; c < 8; ++c) {
        AF a;
        if (c < 4) {
            load_a_row(a, hbf + (long)nc * HID + c*32 + ko);
        } else {
            const float* ap = aggr + (long)nc * HID + (c - 4)*32 + ko;
            float4 f0 = ((const float4*)ap)[0];
            float4 f1 = ((const float4*)ap)[1];
            float4 f2 = ((const float4*)(ap + 16))[0];
            float4 f3 = ((const float4*)(ap + 16))[1];
            a.u[0]=f2bf(f0.x*invd); a.u[1]=f2bf(f0.y*invd); a.u[2]=f2bf(f0.z*invd); a.u[3]=f2bf(f0.w*invd);
            a.u[4]=f2bf(f1.x*invd); a.u[5]=f2bf(f1.y*invd); a.u[6]=f2bf(f1.z*invd); a.u[7]=f2bf(f1.w*invd);
            a.u[8]=f2bf(f2.x*invd); a.u[9]=f2bf(f2.y*invd); a.u[10]=f2bf(f2.z*invd); a.u[11]=f2bf(f2.w*invd);
            a.u[12]=f2bf(f3.x*invd); a.u[13]=f2bf(f3.y*invd); a.u[14]=f2bf(f3.z*invd); a.u[15]=f2bf(f3.w*invd);
        }
        AF b[8];
#pragma unroll
        for (int nt = 0; nt < 8; ++nt)
            load_b(b[nt], s_w1 + (nt*16 + m) * KUPD + c*32 + ko);
#pragma unroll
        for (int nt = 0; nt < 8; ++nt)
            acc[nt] = wmma_bf16(a, b[nt], acc[nt]);
    }

    // bias + relu -> LDS scratch
#pragma unroll
    for (int nt = 0; nt < 8; ++nt) {
        int ncol = nt*16 + m;
        float bias = b1g[ncol];
#pragma unroll
        for (int r = 0; r < 8; ++r) {
            float v = fmaxf(acc[nt][r] + bias, 0.f);
            int row = half ? (r + 8) : r;
            scr[row * HID + ncol] = f2bf(v);
        }
    }

    // GEMM2: (16 x 128) @ (128 x 128)
    v8f acc2[8];
#pragma unroll
    for (int nt = 0; nt < 8; ++nt) acc2[nt] = v8f{0.f,0.f,0.f,0.f,0.f,0.f,0.f,0.f};
#pragma unroll
    for (int c2 = 0; c2 < 4; ++c2) {
        AF a; load_a_row(a, scr + m * HID + c2*32 + ko);
        AF b[8];
#pragma unroll
        for (int nt = 0; nt < 8; ++nt)
            load_b(b[nt], s_w2 + (nt*16 + m) * HID + c2*32 + ko);
#pragma unroll
        for (int nt = 0; nt < 8; ++nt)
            acc2[nt] = wmma_bf16(a, b[nt], acc2[nt]);
    }

    // bias -> bf16 -> scratch, then coalesced row store to h
#pragma unroll
    for (int nt = 0; nt < 8; ++nt) {
        int ncol = nt*16 + m;
        float bias = b2g[ncol];
#pragma unroll
        for (int r = 0; r < 8; ++r) {
            int row = half ? (r + 8) : r;
            scr[row * HID + ncol] = f2bf(acc2[nt][r] + bias);
        }
    }
#pragma unroll
    for (int t = 0; t < 8; ++t) {
        int idx = t * 32 + lane;        // 256 uint4 = 16 rows x 16 uint4
        int row = idx >> 4;
        int col4 = idx & 15;
        int gn = tile * 16 + row;
        if (gn < N)
            ((uint4*)hbf)[(long)gn * 16 + col4] = ((const uint4*)scr)[idx];
    }
}

// ---------------- output MLP + total-energy reduction (VALU; ~1.6 GFLOP) ----------------

__global__ __launch_bounds__(256) void k_output(
    const unsigned short* __restrict__ hbf, const float* __restrict__ w1,
    const float* __restrict__ b1, const float* __restrict__ w2,
    const float* __restrict__ b2, float* __restrict__ out, int N)
{
    __shared__ float s_w1[HID * HID];
    __shared__ float s_b1[HID];
    __shared__ float s_w2[HID];
    for (int t = threadIdx.x; t < HID*HID/4; t += 256)
        ((float4*)s_w1)[t] = ((const float4*)w1)[t];
    if (threadIdx.x < HID) {
        s_b1[threadIdx.x] = b1[threadIdx.x];
        s_w2[threadIdx.x] = w2[threadIdx.x];
    }
    __syncthreads();

    int wave = threadIdx.x >> 5, lane = threadIdx.x & 31;
    int node = blockIdx.x * 8 + wave;
    if (node >= N) return;

    uint4 hr[16];
    const uint4* hp = (const uint4*)(hbf + (long)node * HID);
#pragma unroll
    for (int q = 0; q < 16; ++q) hr[q] = hp[q];

    float tk[4];
#pragma unroll
    for (int k4 = 0; k4 < 4; ++k4) tk[k4] = s_b1[lane + 32*k4];

#pragma unroll
    for (int q = 0; q < 16; ++q) {
        unsigned int cc[4] = { hr[q].x, hr[q].y, hr[q].z, hr[q].w };
#pragma unroll
        for (int ci = 0; ci < 4; ++ci) {
#pragma unroll
            for (int hs = 0; hs < 2; ++hs) {
                int j = q*8 + ci*2 + hs;
                float hj = bf2f((unsigned short)(hs ? (cc[ci] >> 16) : (cc[ci] & 0xFFFFu)));
#pragma unroll
                for (int k4 = 0; k4 < 4; ++k4)
                    tk[k4] += hj * s_w1[j * HID + lane + 32*k4];
            }
        }
    }
    float e = 0.f;
#pragma unroll
    for (int k4 = 0; k4 < 4; ++k4) e += fmaxf(tk[k4], 0.f) * s_w2[lane + 32*k4];
#pragma unroll
    for (int off = 16; off; off >>= 1) e += __shfl_xor(e, off, 32);
    if (lane == 0) atomicAdd(out, e + b2[0]);
}

// ---------------- host launch ----------------

extern "C" void kernel_launch(void* const* d_in, const int* in_sizes, int n_in,
                              void* d_out, int out_size, void* d_ws, size_t ws_size,
                              hipStream_t stream) {
    const int*   z      = (const int*)  d_in[0];
    const float* pos    = (const float*)d_in[1];
    const int*   eidx   = (const int*)  d_in[2];
    const float* emb    = (const float*)d_in[3];
    const float* msg_w1 = (const float*)d_in[4];
    const float* msg_b1 = (const float*)d_in[5];
    const float* msg_w2 = (const float*)d_in[6];
    const float* msg_b2 = (const float*)d_in[7];
    const float* upd_w1 = (const float*)d_in[8];
    const float* upd_b1 = (const float*)d_in[9];
    const float* upd_w2 = (const float*)d_in[10];
    const float* upd_b2 = (const float*)d_in[11];
    const float* out_w1 = (const float*)d_in[12];
    const float* out_b1 = (const float*)d_in[13];
    const float* out_w2 = (const float*)d_in[14];
    const float* out_b2 = (const float*)d_in[15];
    float* out = (float*)d_out;

    int N = in_sizes[0];
    int E = in_sizes[2] / 2;
    const int* ei = eidx;
    const int* ej = eidx + E;

    // workspace partition (256B aligned)
    char* ws = (char*)d_ws;
    size_t o = 0;
    auto take = [&](size_t b) { char* p = ws + o; o += (b + 255) & ~(size_t)255; return p; };
    unsigned short* hbf     = (unsigned short*)take((size_t)N * HID * 2);
    float*          dist    = (float*)take((size_t)E * 4);
    float*          degv    = (float*)take((size_t)N * 4);
    float*          aggr    = (float*)take((size_t)N * HID * 4);
    unsigned short* msg_w1t = (unsigned short*)take((size_t)NLAYERS * KMSG * HID * 2);
    unsigned short* msg_w2t = (unsigned short*)take((size_t)NLAYERS * HID * HID * 2);
    unsigned short* upd_w1t = (unsigned short*)take((size_t)NLAYERS * KUPD * HID * 2);
    unsigned short* upd_w2t = (unsigned short*)take((size_t)NLAYERS * HID * HID * 2);

    k_zero<<<1, 256, 0, stream>>>(out, (long)out_size);
    k_zero<<<256, 256, 0, stream>>>(degv, (long)N);
    k_embed<<<(int)(((long)N * HID + 255) / 256), 256, 0, stream>>>(z, emb, hbf, N);
    k_wt<<<(int)(((long)NLAYERS*KMSG*HID + 255) / 256), 256, 0, stream>>>(msg_w1, msg_w1t, KMSG, NLAYERS);
    k_wt<<<(int)(((long)NLAYERS*HID*HID  + 255) / 256), 256, 0, stream>>>(msg_w2, msg_w2t, HID, NLAYERS);
    k_wt<<<(int)(((long)NLAYERS*KUPD*HID + 255) / 256), 256, 0, stream>>>(upd_w1, upd_w1t, KUPD, NLAYERS);
    k_wt<<<(int)(((long)NLAYERS*HID*HID  + 255) / 256), 256, 0, stream>>>(upd_w2, upd_w2t, HID, NLAYERS);
    k_edge<<<(E + 255) / 256, 256, 0, stream>>>(pos, ei, ej, dist, degv, E);
    k_invdeg<<<(N + 255) / 256, 256, 0, stream>>>(degv, N);

    int mTiles = (E + 15) / 16, mBlocks = (mTiles + 7) / 8;
    int uTiles = (N + 15) / 16, uBlocks = (uTiles + 7) / 8;

    for (int l = 0; l < NLAYERS; ++l) {
        k_zero<<<4096, 256, 0, stream>>>(aggr, (long)N * HID);
        k_message<<<mBlocks, 256, 0, stream>>>(
            hbf, dist, ei, ej,
            msg_w1t + (size_t)l * KMSG * HID, msg_w2t + (size_t)l * HID * HID,
            msg_b1 + l * HID, msg_b2 + l * HID, aggr, E);
        k_update<<<uBlocks, 256, 0, stream>>>(
            hbf, aggr, degv,
            upd_w1t + (size_t)l * KUPD * HID, upd_w2t + (size_t)l * HID * HID,
            upd_b1 + l * HID, upd_b2 + l * HID, N);
    }
    k_output<<<(N + 7) / 8, 256, 0, stream>>>(hbf, out_w1, out_b1, out_w2, out_b2, out, N);
}
